// DrugPropertyGNN_11931419148847
// MI455X (gfx1250) — compile-verified, matrix operand order checked by reading, not run on hardware
//
#include <hip/hip_runtime.h>

#define NN 100000
#define NG 4096
#define IN_DIM 75
#define HID 64

typedef __attribute__((ext_vector_type(16))) _Float16 v16h;
typedef __attribute__((ext_vector_type(8)))  _Float16 v8h;
typedef __attribute__((ext_vector_type(8)))  float    v8f;

// ---------------- utility fill ----------------
__global__ void fill_f32(float* __restrict__ p, float v, int n) {
    int i = blockIdx.x * blockDim.x + threadIdx.x;
    if (i < n) p[i] = v;
}

// ---------------- degree / rsqrt / graph counts ----------------
__global__ void degree_kernel(const int* __restrict__ dst, float* __restrict__ deg, int e) {
    int i = blockIdx.x * blockDim.x + threadIdx.x;
    if (i < e) unsafeAtomicAdd(&deg[dst[i]], 1.0f);
}

__global__ void rsqrt_kernel(float* __restrict__ deg, int n) {
    int i = blockIdx.x * blockDim.x + threadIdx.x;
    if (i < n) deg[i] = rsqrtf(deg[i]);
}

__global__ void counts_kernel(const int* __restrict__ batch, float* __restrict__ counts, int n) {
    int i = blockIdx.x * blockDim.x + threadIdx.x;
    if (i < n) unsafeAtomicAdd(&counts[batch[i]], 1.0f);
}

// ---------------- pack: f32 [rows,K] -> f16 [rows,Kp] zero-padded ----------------
__global__ void pack_a_kernel(const float* __restrict__ x, _Float16* __restrict__ Ap,
                              int K, int Kp, int n) {
    int idx = blockIdx.x * blockDim.x + threadIdx.x;
    if (idx >= n) return;
    int row = idx / Kp;
    int k   = idx - row * Kp;
    Ap[idx] = (k < K) ? (_Float16)x[(size_t)row * K + k] : (_Float16)0.0f;
}

// ---------------- pack: W f32 [K,64] -> Wt f16 [64,Kp] (column-major, padded) ----------------
__global__ void pack_w_kernel(const float* __restrict__ W, _Float16* __restrict__ Wt,
                              int K, int Kp, int n) {
    int idx = blockIdx.x * blockDim.x + threadIdx.x;
    if (idx >= n) return;
    int ncol = idx / Kp;
    int k    = idx - ncol * Kp;
    Wt[idx] = (k < K) ? (_Float16)W[(size_t)k * HID + ncol] : (_Float16)0.0f;
}

// ---------------- WMMA GEMM: O[nrows,64] = Ap[nrows,Kp] @ Wt^T ----------------
// Ap: f16 row-major padded; Wt: f16 [64][Kp] column(n)-major. Kp % 32 == 0.
// One wave32 per 16-row tile; fragments load as two contiguous 16-byte chunks per lane.
__global__ __launch_bounds__(128) void gemm_wmma_kernel(
        const _Float16* __restrict__ Ap, int Kp,
        const _Float16* __restrict__ Wt,
        float* __restrict__ O, int nrows) {
    const int wave = threadIdx.x >> 5;
    const int lane = threadIdx.x & 31;
    const int tile = blockIdx.x * 4 + wave;
    if (tile >= (nrows >> 4)) return;         // wave-uniform: EXEC stays all-ones
    const int row0 = tile << 4;
    const int hs   = lane >> 4;               // half-wave select
    const int mr   = lane & 15;               // A-frag row / B-frag column

    v8f acc[4];
#pragma unroll
    for (int t = 0; t < 4; ++t)
#pragma unroll
        for (int r = 0; r < 8; ++r) acc[t][r] = 0.0f;

    const _Float16* ap = Ap + (size_t)(row0 + mr) * Kp + (hs << 3);
    const _Float16* bp0 = Wt + (size_t)mr * Kp + (hs << 3);

    for (int k0 = 0; k0 < Kp; k0 += 32) {
        // A fragment: K = {k0+hs*8 .. +7} in elems 0-7, {k0+16+hs*8 .. +7} in elems 8-15
        v8h alo = *(const v8h*)(ap + k0);
        v8h ahi = *(const v8h*)(ap + k0 + 16);
        v16h af = __builtin_shufflevector(alo, ahi,
                    0, 1, 2, 3, 4, 5, 6, 7, 8, 9, 10, 11, 12, 13, 14, 15);
#pragma unroll
        for (int t = 0; t < 4; ++t) {
            const _Float16* bp = bp0 + (size_t)(t * 16) * Kp + k0;
            v8h blo = *(const v8h*)(bp);
            v8h bhi = *(const v8h*)(bp + 16);
            v16h bf = __builtin_shufflevector(blo, bhi,
                        0, 1, 2, 3, 4, 5, 6, 7, 8, 9, 10, 11, 12, 13, 14, 15);
            acc[t] = __builtin_amdgcn_wmma_f32_16x16x32_f16(
                false, af, false, bf, (short)0, acc[t], false, false);
        }
    }

    // C/D layout: VGPR r -> M = r + 8*hs, N = mr
#pragma unroll
    for (int t = 0; t < 4; ++t) {
        const int n = t * 16 + mr;
#pragma unroll
        for (int r = 0; r < 8; ++r) {
            const int m = row0 + r + (hs << 3);
            O[(size_t)m * HID + n] = acc[t][r];
        }
    }
}

// ---------------- edge scatter: Acc[dst] += dinv[src]*dinv[dst] * X[src] ----------------
// One wave32 per edge, 2 channels per lane. Non-returning f32 atomics resolve in L2
// (25.6 MB destination buffer fits in the 192 MB L2).
__global__ void scatter_kernel(const int* __restrict__ src, const int* __restrict__ dst,
                               const float* __restrict__ dinv,
                               const float* __restrict__ X, float* __restrict__ Acc, int E) {
    const int gw   = (blockIdx.x * blockDim.x + threadIdx.x) >> 5;
    const int lane = threadIdx.x & 31;
    if (gw >= E) return;
    const int s = src[gw];
    const int d = dst[gw];
    const float nrm = dinv[s] * dinv[d];
    const float2 v = ((const float2*)(X + (size_t)s * HID))[lane];
    float* ad = Acc + (size_t)d * HID + lane * 2;
    unsafeAtomicAdd(ad,     v.x * nrm);
    unsafeAtomicAdd(ad + 1, v.y * nrm);
}

// ---------------- layer-1 finalize: Hh = f16(relu(Acc + dinv^2*X + b)); Acc = 0 ----------------
__global__ void finalize_h16_kernel(const float* __restrict__ Acc, const float* __restrict__ X,
                                    const float* __restrict__ dinv, const float* __restrict__ b,
                                    _Float16* __restrict__ Hh, float* __restrict__ AccReset, int n) {
    int idx = blockIdx.x * blockDim.x + threadIdx.x;
    if (idx >= n) return;
    const int i = idx >> 6;
    const int c = idx & 63;
    const float di = dinv[i];
    const float v = Acc[idx] + di * di * X[idx] + b[c];
    Hh[idx] = (_Float16)fmaxf(v, 0.0f);
    AccReset[idx] = 0.0f;
}

// ---------------- layer-2 finalize fused with mean-pool accumulation ----------------
__global__ void finalize_pool_kernel(const float* __restrict__ Acc, const float* __restrict__ X,
                                     const float* __restrict__ dinv, const float* __restrict__ b,
                                     const int* __restrict__ batch,
                                     float* __restrict__ sums, int n) {
    int idx = blockIdx.x * blockDim.x + threadIdx.x;
    if (idx >= n) return;
    const int i = idx >> 6;
    const int c = idx & 63;
    const float di = dinv[i];
    const float v = fmaxf(Acc[idx] + di * di * X[idx] + b[c], 0.0f);
    unsafeAtomicAdd(&sums[(size_t)batch[i] * HID + c], v);
}

// ---------------- head: out[g] = (sums[g]/max(cnt,1)) . Wfc + bfc ----------------
__global__ void head_kernel(const float* __restrict__ sums, const float* __restrict__ counts,
                            const float* __restrict__ Wfc, const float* __restrict__ bfc,
                            float* __restrict__ out, int ngraphs) {
    int g = blockIdx.x * blockDim.x + threadIdx.x;
    if (g >= ngraphs) return;
    const float inv = 1.0f / fmaxf(counts[g], 1.0f);
    float acc = 0.0f;
#pragma unroll
    for (int c = 0; c < HID; ++c) acc += sums[(size_t)g * HID + c] * Wfc[c];
    out[g] = acc * inv + bfc[0];
}

extern "C" void kernel_launch(void* const* d_in, const int* in_sizes, int n_in,
                              void* d_out, int out_size, void* d_ws, size_t ws_size,
                              hipStream_t stream) {
    const float* x     = (const float*)d_in[0];
    const int*   ei    = (const int*)  d_in[1];   // [2, E] flattened
    const int*   batch = (const int*)  d_in[2];
    const float* W1    = (const float*)d_in[3];
    const float* b1    = (const float*)d_in[4];
    const float* W2    = (const float*)d_in[5];
    const float* b2    = (const float*)d_in[6];
    const float* Wfc   = (const float*)d_in[7];
    const float* bfc   = (const float*)d_in[8];
    float*       out   = (float*)d_out;

    const int E = in_sizes[1] / 2;
    const int* src = ei;
    const int* dst = ei + E;

    const int KP1 = 96;   // IN_DIM=75 padded to multiple of 32
    const int KP2 = 64;   // HID already multiple of 32
    const int NF  = NN * HID;

    // workspace layout: f32 regions first, then f16 regions (all 256-float aligned)
    float* ws     = (float*)d_ws;
    float* deg    = ws;                               // NN f32 (deg -> dinv in place)
    float* X      = deg + 100096;                     // NN*HID f32 (xw / hw buffer)
    float* Acc    = X   + (size_t)NN * HID;           // NN*HID f32 (edge accumulator)
    float* sums   = Acc + (size_t)NN * HID;           // NG*HID f32
    float* counts = sums + (size_t)NG * HID;          // NG f32
    _Float16* Ap  = (_Float16*)(counts + NG + 256);   // NN*KP1 f16 (packed x)
    _Float16* Hh  = Ap + (size_t)NN * KP1;            // NN*HID f16 (layer-1 activations)
    _Float16* Wt1 = Hh + (size_t)NN * HID;            // 64*KP1 f16 (W1^T padded)
    _Float16* Wt2 = Wt1 + (size_t)HID * KP1;          // 64*KP2 f16 (W2^T)

    // init (workspace not re-poisoned between timed replays: re-zero every call)
    fill_f32<<<(NN + 255) / 256, 256, 0, stream>>>(deg, 1.0f, NN);   // self-loop pre-count
    fill_f32<<<(NF + 255) / 256, 256, 0, stream>>>(Acc, 0.0f, NF);
    fill_f32<<<(NG * HID + 255) / 256, 256, 0, stream>>>(sums, 0.0f, NG * HID);
    fill_f32<<<(NG + 255) / 256, 256, 0, stream>>>(counts, 0.0f, NG);

    // normalization + graph sizes
    degree_kernel<<<(E + 255) / 256, 256, 0, stream>>>(dst, deg, E);
    rsqrt_kernel<<<(NN + 255) / 256, 256, 0, stream>>>(deg, NN);
    counts_kernel<<<(NN + 255) / 256, 256, 0, stream>>>(batch, counts, NN);

    // operand packing (f32 -> padded f16; weights transposed to column-major)
    pack_a_kernel<<<(NN * KP1 + 255) / 256, 256, 0, stream>>>(x, Ap, IN_DIM, KP1, NN * KP1);
    pack_w_kernel<<<(HID * KP1 + 255) / 256, 256, 0, stream>>>(W1, Wt1, IN_DIM, KP1, HID * KP1);
    pack_w_kernel<<<(HID * KP2 + 255) / 256, 256, 0, stream>>>(W2, Wt2, HID,    KP2, HID * KP2);

    const int gemm_blocks = ((NN >> 4) + 3) / 4;   // 4 waves (tiles) per block
    const int edge_blocks = (E * 32 + 255) / 256;  // one wave per edge

    // ---- layer 1 ----
    gemm_wmma_kernel<<<gemm_blocks, 128, 0, stream>>>(Ap, KP1, Wt1, X, NN);
    scatter_kernel<<<edge_blocks, 256, 0, stream>>>(src, dst, deg, X, Acc, E);
    finalize_h16_kernel<<<(NF + 255) / 256, 256, 0, stream>>>(Acc, X, deg, b1, Hh, Acc, NF);

    // ---- layer 2 ----
    gemm_wmma_kernel<<<gemm_blocks, 128, 0, stream>>>(Hh, KP2, Wt2, X, NN);
    scatter_kernel<<<edge_blocks, 256, 0, stream>>>(src, dst, deg, X, Acc, E);
    finalize_pool_kernel<<<(NF + 255) / 256, 256, 0, stream>>>(Acc, X, deg, b2, batch, sums, NF);

    // ---- head ----
    head_kernel<<<(NG + 255) / 256, 256, 0, stream>>>(sums, counts, Wfc, bfc, out, NG);
}
